// ImageZoomxModel_70617852281429
// MI455X (gfx1250) — compile-verified, matrix operand order checked by reading, not run on hardware
//
#include <hip/hip_runtime.h>
#include <hip/hip_bf16.h>

typedef __attribute__((ext_vector_type(16))) _Float16 v16h;
typedef __attribute__((ext_vector_type(8)))  _Float16 v8h;
typedef __attribute__((ext_vector_type(8)))  float    v8f;

#define IN_HW   128
#define NPIX    16384       // 128*128
#define OUT_HW  512
#define NOUT    262144      // 512*512
#define FEAT_C  64
#define K_CONV  576         // 64*9
#define K_MLP0  608         // 580 padded to 19*32
#define HID     256

// ---------------------------------------------------------------------------
// WMMA fragment loaders.
// 16-bit A 16x32 layout (ISA 7.12.2): lane L holds row M=L%16; low half-lanes
// hold K {0..7, 16..23}, high half-lanes K {8..15, 24..31}: two contiguous
// 16-byte chunks, 32 bytes apart. B from transposed weights Wt[N][Kpad] uses
// the mirrored pattern (lane -> column N).
// ---------------------------------------------------------------------------
__device__ __forceinline__ v16h ldfragA(const _Float16* rows, int stride, int k0) {
    int lane = threadIdx.x & 31;
    int m = lane & 15, hi = lane >> 4;
    const _Float16* p = rows + m * stride + k0 + hi * 8;
    v8h a0 = *(const v8h*)(p);
    v8h a1 = *(const v8h*)(p + 16);
    v16h r;
#pragma unroll
    for (int i = 0; i < 8; i++) { r[i] = a0[i]; r[i + 8] = a1[i]; }
    return r;
}

__device__ __forceinline__ v16h ldfragB(const _Float16* Wt, int Kpad, int n0, int k0) {
    int lane = threadIdx.x & 31;
    int n = n0 + (lane & 15), hi = lane >> 4;
    const _Float16* p = Wt + (size_t)n * Kpad + k0 + hi * 8;
    v8h b0 = *(const v8h*)(p);
    v8h b1 = *(const v8h*)(p + 16);
    v16h r;
#pragma unroll
    for (int i = 0; i < 8; i++) { r[i] = b0[i]; r[i + 8] = b1[i]; }
    return r;
}

// ---------------------------------------------------------------------------
// Weight prep kernels
// ---------------------------------------------------------------------------
__global__ void f32_to_f16(const float* __restrict__ s, _Float16* __restrict__ d, int n) {
    int i = blockIdx.x * blockDim.x + threadIdx.x;
    if (i < n) d[i] = (_Float16)s[i];
}

// dst[Nout][Kpad] = src[K][N]^T, zero padded
__global__ void transpose_pad(const float* __restrict__ src, _Float16* __restrict__ dst,
                              int K, int N, int Kpad, int Nout) {
    int i = blockIdx.x * blockDim.x + threadIdx.x;
    int total = Nout * Kpad;
    if (i >= total) return;
    int n = i / Kpad, k = i - n * Kpad;
    float v = (n < N && k < K) ? src[k * N + n] : 0.0f;
    dst[i] = (_Float16)v;
}

// ---------------------------------------------------------------------------
// Head conv: 3 -> 64 channels, plain FMA (56 MFLOP, negligible)
// ---------------------------------------------------------------------------
__global__ void head_conv(const float* __restrict__ x, const float* __restrict__ wgt,
                          const float* __restrict__ b, float* __restrict__ h0) {
    int idx = blockIdx.x * blockDim.x + threadIdx.x;
    if (idx >= FEAT_C * NPIX) return;
    int oc = idx >> 14, pix = idx & (NPIX - 1);
    int y = pix >> 7, x0 = pix & 127;
    float s = b[oc];
#pragma unroll
    for (int ic = 0; ic < 3; ic++)
#pragma unroll
        for (int kh = 0; kh < 3; kh++)
#pragma unroll
            for (int kw = 0; kw < 3; kw++) {
                int yy = y + kh - 1, xx = x0 + kw - 1;
                if ((unsigned)yy < 128u && (unsigned)xx < 128u)
                    s += x[ic * NPIX + yy * 128 + xx] * wgt[((oc * 3 + ic) * 3 + kh) * 3 + kw];
            }
    h0[idx] = s;
}

// ---------------------------------------------------------------------------
// 64->64 3x3 conv as implicit GEMM via WMMA.
// Wave tile: 16 pixels x 64 out-channels. A = im2col [16 x 576] fp16 in LDS.
// K-outer / N-inner loop nest: each A fragment is loaded once per K-step and
// reused across 4 accumulator tiles held in registers. RELU / HAS_RES are
// template params so the epilogue has no per-element branches.
// ---------------------------------------------------------------------------
#define CONV_WAVES 4
template <bool RELU, bool HAS_RES>
__global__ void __launch_bounds__(CONV_WAVES * 32, 1)
conv64_wmma(const float* __restrict__ in, const _Float16* __restrict__ Wt,
            const float* __restrict__ bias, const float* __restrict__ res,
            float* __restrict__ out) {
    __shared__ __align__(32) _Float16 smem[CONV_WAVES * 16 * K_CONV];
    int wave = threadIdx.x >> 5, lane = threadIdx.x & 31;
    _Float16* A = smem + wave * (16 * K_CONV);
    int tile = blockIdx.x * CONV_WAVES + wave;      // 0..1023
    int p0 = tile * 16;
    int y = p0 >> 7, x0 = p0 & 127;

    // build im2col tile (fp32 -> fp16)
#pragma unroll 1
    for (int m = 0; m < 16; m++) {
        int xm = x0 + m;
        for (int k = lane; k < K_CONV; k += 32) {
            int c = k / 9, t = k - c * 9, kh = t / 3, kw = t - kh * 3;
            int yy = y + kh - 1, xx = xm + kw - 1;
            float v = 0.0f;
            if ((unsigned)yy < 128u && (unsigned)xx < 128u)
                v = in[c * NPIX + yy * 128 + xx];
            A[m * K_CONV + k] = (_Float16)v;
        }
    }
    __syncthreads();

    int nl = lane & 15, hi = lane >> 4;
    v8f acc[4];
#pragma unroll
    for (int nt = 0; nt < 4; nt++) acc[nt] = (v8f){};

#pragma unroll 1
    for (int ks = 0; ks < 18; ks++) {
        v16h a = ldfragA(A, K_CONV, ks * 32);
#pragma unroll
        for (int nt = 0; nt < 4; nt++) {
            v16h b = ldfragB(Wt, K_CONV, nt * 16, ks * 32);
            acc[nt] = __builtin_amdgcn_wmma_f32_16x16x32_f16(false, a, false, b,
                                                             (short)0, acc[nt], false, false);
        }
    }

#pragma unroll
    for (int nt = 0; nt < 4; nt++) {
        int n = nt * 16 + nl;
        float bv = bias[n];
#pragma unroll
        for (int r = 0; r < 8; r++) {
            int m = r + hi * 8;
            int pix = p0 + m;
            float v = acc[nt][r] + bv;
            if (RELU) v = fmaxf(v, 0.0f);
            if (HAS_RES) v += res[n * NPIX + pix];
            out[n * NPIX + pix] = v;
        }
    }
}

// ---------------------------------------------------------------------------
// LIIF query kernel. One wave = 16 output pixels, full 5-layer MLP per
// ensemble sample via WMMA, diagonal-area ensemble accumulation, clip, store.
// Per-wave LDS arena: A[16x608] (aliased by hidden ping buffer) + H0[16x256].
// ---------------------------------------------------------------------------
#define Q_WAVES 4
struct __align__(32) QWave {
    _Float16 A[16 * K_MLP0];   // 19456 B ; first 8192 B reused as H1
    _Float16 H0[16 * HID];     // 8192 B
    float area[4][16];
    float relh[4][16];
    float relw[4][16];
    int   iy[4][16];
    int   ix[4][16];
};

// K-outer / N-inner: 16 accumulator tiles (128 VGPRs) live in registers,
// one A fragment per K-step shared across all 16 WMMA chains.
template <int KSTEPS>
__device__ __forceinline__ void mlp_layer(const _Float16* Ain, int strideA,
                                          const _Float16* __restrict__ Wt, int Kpad,
                                          const float* __restrict__ bias,
                                          _Float16* Hout) {
    int lane = threadIdx.x & 31;
    int nl = lane & 15, hi = lane >> 4;
    v8f acc[16];
#pragma unroll
    for (int nt = 0; nt < 16; nt++) acc[nt] = (v8f){};

#pragma unroll 1
    for (int ks = 0; ks < KSTEPS; ks++) {
        v16h a = ldfragA(Ain, strideA, ks * 32);
#pragma unroll
        for (int nt = 0; nt < 16; nt++) {
            v16h b = ldfragB(Wt, Kpad, nt * 16, ks * 32);
            acc[nt] = __builtin_amdgcn_wmma_f32_16x16x32_f16(false, a, false, b,
                                                             (short)0, acc[nt], false, false);
        }
    }

#pragma unroll
    for (int nt = 0; nt < 16; nt++) {
        int n = nt * 16 + nl;
        float bv = bias[n];
#pragma unroll
        for (int r = 0; r < 8; r++) {
            int m = r + hi * 8;
            float v = fmaxf(acc[nt][r] + bv, 0.0f);   // relu (layers 1-4 only)
            Hout[m * HID + n] = (_Float16)v;
        }
    }
}

__global__ void __launch_bounds__(Q_WAVES * 32, 1)
liif_query(const _Float16* __restrict__ feat16,
           const _Float16* __restrict__ W0t, const float* __restrict__ b0,
           const _Float16* __restrict__ W1t, const float* __restrict__ b1,
           const _Float16* __restrict__ W2t, const float* __restrict__ b2,
           const _Float16* __restrict__ W3t, const float* __restrict__ b3,
           const _Float16* __restrict__ W4t, const float* __restrict__ b4,
           float* __restrict__ outp) {
    __shared__ QWave sm[Q_WAVES];
    int wave = threadIdx.x >> 5, lane = threadIdx.x & 31;
    QWave& w = sm[wave];
    _Float16* A  = w.A;
    _Float16* H0 = w.H0;
    _Float16* H1 = w.A;                 // alias: A dead after layer 1
    int tile = blockIdx.x * Q_WAVES + wave;
    int q0 = tile * 16;
    int nl = lane & 15, hi = lane >> 4;

    // per-query, per-sample geometry (lanes 0..15, one query each)
    if (lane < 16) {
        int q = q0 + lane;
        int oy = q >> 9, ox = q & 511;
        float sh = (oy + 0.5f) * (2.0f / OUT_HW) - 1.0f;
        float sw = (ox + 0.5f) * (2.0f / OUT_HW) - 1.0f;
        const float d = 1.0f / 128.0f, EPSc = 1e-6f;
#pragma unroll
        for (int s = 0; s < 4; s++) {
            float gh = fminf(fmaxf(sh + ((s >> 1) ? d : -d), -1.0f + EPSc), 1.0f - EPSc);
            float gw = fminf(fmaxf(sw + ((s & 1) ? d : -d), -1.0f + EPSc), 1.0f - EPSc);
            // torch nearest grid_sample, align_corners=False, round-half-even
            int iy = (int)rintf((gh + 1.0f) * 64.0f - 0.5f);
            int ix = (int)rintf((gw + 1.0f) * 64.0f - 0.5f);
            iy = min(max(iy, 0), 127);
            ix = min(max(ix, 0), 127);
            float qgh = (iy + 0.5f) * (2.0f / 128.0f) - 1.0f;
            float qgw = (ix + 0.5f) * (2.0f / 128.0f) - 1.0f;
            float rh = (sh - qgh) * 128.0f;
            float rw = (sw - qgw) * 128.0f;
            w.iy[s][lane] = iy;  w.ix[s][lane] = ix;
            w.relh[s][lane] = rh; w.relw[s][lane] = rw;
            w.area[s][lane] = fabsf(rh * rw);
        }
    }
    __syncthreads();

    float acc[8];
#pragma unroll
    for (int r = 0; r < 8; r++) acc[r] = 0.0f;

#pragma unroll 1
    for (int s = 0; s < 4; s++) {
        // gather 576 im2col features per query row into A (fp16, L2-hot map)
#pragma unroll 1
        for (int m = 0; m < 16; m++) {
            int iy = w.iy[s][m], ix = w.ix[s][m];
            for (int k = lane; k < 576; k += 32) {
                int c = k / 9, t = k - c * 9, kh = t / 3, kw = t - kh * 3;
                int yy = iy + kh - 1, xx = ix + kw - 1;
                _Float16 v = (_Float16)0.0f;
                if ((unsigned)yy < 128u && (unsigned)xx < 128u)
                    v = feat16[c * NPIX + yy * 128 + xx];
                A[m * K_MLP0 + k] = v;
            }
        }
        if (lane < 16) {
            int m = lane;
            A[m * K_MLP0 + 576] = (_Float16)w.relh[s][m];
            A[m * K_MLP0 + 577] = (_Float16)w.relw[s][m];
            A[m * K_MLP0 + 578] = (_Float16)0.5f;   // q_cell h = 2*128/512
            A[m * K_MLP0 + 579] = (_Float16)0.5f;   // q_cell w
            for (int k = 580; k < K_MLP0; k++) A[m * K_MLP0 + k] = (_Float16)0.0f;
        }
        __syncthreads();

        mlp_layer<19>(A,  K_MLP0, W0t, K_MLP0, b0, H0); __syncthreads();
        mlp_layer<8 >(H0, HID,    W1t, HID,    b1, H1); __syncthreads();
        mlp_layer<8 >(H1, HID,    W2t, HID,    b2, H0); __syncthreads();
        mlp_layer<8 >(H0, HID,    W3t, HID,    b3, H1); __syncthreads();

        // final layer 256 -> 3 (N padded to 16), kept in registers
        v8f cf = {};
#pragma unroll
        for (int ks = 0; ks < 8; ks++) {
            v16h a = ldfragA(H1, HID, ks * 32);
            v16h b = ldfragB(W4t, HID, 0, ks * 32);
            cf = __builtin_amdgcn_wmma_f32_16x16x32_f16(false, a, false, b,
                                                        (short)0, cf, false, false);
        }
        float bv = (nl < 3) ? b4[nl] : 0.0f;
#pragma unroll
        for (int r = 0; r < 8; r++) {
            int m = r + hi * 8;
            acc[r] += (cf[r] + bv) * w.area[3 - s][m];   // diagonal-swapped areas
        }
        __syncthreads();
    }

    if (nl < 3) {
#pragma unroll
        for (int r = 0; r < 8; r++) {
            int m = r + hi * 8;
            float den = w.area[0][m] + w.area[1][m] + w.area[2][m] + w.area[3][m] + 1e-9f;
            float v = acc[r] / den;
            v = fminf(fmaxf(v, 0.0f), 1.0f);
            outp[nl * NOUT + (q0 + m)] = v;
        }
    }
}

// ---------------------------------------------------------------------------
// Orchestration
// ---------------------------------------------------------------------------
extern "C" void kernel_launch(void* const* d_in, const int* in_sizes, int n_in,
                              void* d_out, int out_size, void* d_ws, size_t ws_size,
                              hipStream_t stream) {
    (void)in_sizes; (void)n_in; (void)out_size; (void)ws_size;
    const float* x      = (const float*)d_in[0];
    const float* head_w = (const float*)d_in[2];
    const float* head_b = (const float*)d_in[3];
    const float* rb_w   = (const float*)d_in[4];
    const float* rb_b   = (const float*)d_in[5];
    const float* tail_w = (const float*)d_in[6];
    const float* tail_b = (const float*)d_in[7];
    const float* mw0 = (const float*)d_in[8];  const float* mb0 = (const float*)d_in[9];
    const float* mw1 = (const float*)d_in[10]; const float* mb1 = (const float*)d_in[11];
    const float* mw2 = (const float*)d_in[12]; const float* mb2 = (const float*)d_in[13];
    const float* mw3 = (const float*)d_in[14]; const float* mb3 = (const float*)d_in[15];
    const float* mw4 = (const float*)d_in[16]; const float* mb4 = (const float*)d_in[17];
    float* outp = (float*)d_out;

    char* ws = (char*)d_ws;
    const size_t FEAT_BYTES = (size_t)FEAT_C * NPIX * 4;            // 4 MB
    float*    h0     = (float*)(ws);
    float*    hA     = (float*)(ws + FEAT_BYTES);
    float*    hB     = (float*)(ws + 2 * FEAT_BYTES);
    float*    feat64 = (float*)(ws + 3 * FEAT_BYTES);
    _Float16* feat16 = (_Float16*)(ws + 4 * FEAT_BYTES);            // 2 MB
    _Float16* convW  = (_Float16*)(ws + 4 * FEAT_BYTES + 2097152);  // 33 * 36864 fp16
    char*     wbase  = ws + 4 * FEAT_BYTES + 2097152 + 2433024;
    _Float16* W0t = (_Float16*)(wbase);
    _Float16* W1t = (_Float16*)(wbase + 311296);
    _Float16* W2t = (_Float16*)(wbase + 311296 + 131072);
    _Float16* W3t = (_Float16*)(wbase + 311296 + 2 * 131072);
    _Float16* W4t = (_Float16*)(wbase + 311296 + 3 * 131072);

    // --- weight prep ---
    {
        int n = 16 * 2 * FEAT_C * K_CONV;       // 1,179,648
        f32_to_f16<<<(n + 255) / 256, 256, 0, stream>>>(rb_w, convW, n);
        int nt = FEAT_C * K_CONV;               // 36,864
        f32_to_f16<<<(nt + 255) / 256, 256, 0, stream>>>(tail_w, convW + (size_t)32 * 36864, nt);
        int t0 = 256 * K_MLP0;
        transpose_pad<<<(t0 + 255) / 256, 256, 0, stream>>>(mw0, W0t, 580, 256, K_MLP0, 256);
        int th = 256 * 256;
        transpose_pad<<<(th + 255) / 256, 256, 0, stream>>>(mw1, W1t, 256, 256, 256, 256);
        transpose_pad<<<(th + 255) / 256, 256, 0, stream>>>(mw2, W2t, 256, 256, 256, 256);
        transpose_pad<<<(th + 255) / 256, 256, 0, stream>>>(mw3, W3t, 256, 256, 256, 256);
        int t4 = 16 * 256;
        transpose_pad<<<(t4 + 255) / 256, 256, 0, stream>>>(mw4, W4t, 256, 3, 256, 16);
    }

    // --- encoder ---
    head_conv<<<(FEAT_C * NPIX + 255) / 256, 256, 0, stream>>>(x, head_w, head_b, h0);
    hipMemcpyAsync(hA, h0, FEAT_BYTES, hipMemcpyDeviceToDevice, stream);

    const int convBlocks = (NPIX / 16) / CONV_WAVES;   // 256
    for (int b = 0; b < 16; b++) {
        const _Float16* w0 = convW + (size_t)(b * 2 + 0) * 36864;
        const _Float16* w1 = convW + (size_t)(b * 2 + 1) * 36864;
        const float* bb0 = rb_b + (b * 2 + 0) * 64;
        const float* bb1 = rb_b + (b * 2 + 1) * 64;
        conv64_wmma<true,  false><<<convBlocks, CONV_WAVES * 32, 0, stream>>>(hA, w0, bb0, nullptr, hB);
        conv64_wmma<false, true ><<<convBlocks, CONV_WAVES * 32, 0, stream>>>(hB, w1, bb1, hA, hA);
    }
    conv64_wmma<false, true><<<convBlocks, CONV_WAVES * 32, 0, stream>>>(
        hA, convW + (size_t)32 * 36864, tail_b, h0, feat64);

    f32_to_f16<<<(FEAT_C * NPIX + 255) / 256, 256, 0, stream>>>(feat64, feat16, FEAT_C * NPIX);

    // --- LIIF query / MLP ---
    const int qBlocks = (NOUT / 16) / Q_WAVES;         // 4096
    liif_query<<<qBlocks, Q_WAVES * 32, 0, stream>>>(feat16,
        W0t, mb0, W1t, mb1, W2t, mb2, W3t, mb3, W4t, mb4, outp);
}
